// AttnDecoderBlock_4535485465032
// MI455X (gfx1250) — compile-verified, matrix operand order checked by reading, not run on hardware
//
#include <hip/hip_runtime.h>
#include <stdint.h>

typedef __attribute__((ext_vector_type(16))) _Float16 v16h;
typedef __attribute__((ext_vector_type(8)))  _Float16 v8h;
typedef __attribute__((ext_vector_type(8)))  float    v8f;

#define BB    128
#define CC    384
#define NHH   6
#define HD    64
#define WS    20
#define HH    15
#define WW    20
#define LTOK  300          // H*W valid tokens
#define NTOK  400          // padded window tokens
#define NTOKP 416          // NTOK padded to multiple of 32 for WMMA K
#define OUTD  192
#define OH    30
#define OW    40

// ---------------- WMMA fragment helpers ----------------
// A fragment (16x32 f16, row-major source [row][K]):
//   lane&15 = row; lanes 0-15 take K = koff+0..7 and koff+16..23 with koff = 0
//   lanes 16-31 same with koff = 8.  Elements 0..7 -> p[0..7], 8..15 -> p[16..23].
__device__ __forceinline__ v16h load_frag16(const _Float16* p) {
    union { v16h v; v8h h[2]; } u;
    u.h[0] = *(const v8h*)(p);
    u.h[1] = *(const v8h*)(p + 16);
    return u.v;
}

// N-blocked k-loop: one A fragment, NB B fragments / accumulators per step.
template <int NB>
__device__ __forceinline__ void gemm_k_loop_nb(const _Float16* __restrict__ pa,
                                               const _Float16* __restrict__ pb,
                                               long bstride, int K, v8f acc[NB]) {
    for (int k = 0; k < K; k += 32) {
        v16h a = load_frag16(pa + k);
        #pragma unroll
        for (int j = 0; j < NB; ++j) {
            v16h b = load_frag16(pb + (long)j * bstride + k);
            acc[j] = __builtin_amdgcn_wmma_f32_16x16x32_f16(false, a, false, b,
                                                            (short)0, acc[j], false, false);
        }
    }
}

__device__ __forceinline__ v8f gemm_k_loop(const _Float16* __restrict__ pa,
                                           const _Float16* __restrict__ pb,
                                           int K, v8f acc) {
    for (int k = 0; k < K; k += 32) {
        v16h a = load_frag16(pa + k);
        v16h b = load_frag16(pb + k);
        acc = __builtin_amdgcn_wmma_f32_16x16x32_f16(false, a, false, b,
                                                     (short)0, acc, false, false);
    }
    return acc;
}

__device__ __forceinline__ float wave_sum(float v) {
    #pragma unroll
    for (int o = 16; o > 0; o >>= 1) v += __shfl_xor(v, o, 32);
    return v;
}

// ---------------- weight convert: W[K][N] f32 -> Wt[N][K] f16 ----------------
__global__ void wconv_kernel(const float* __restrict__ W, _Float16* __restrict__ Wt,
                             int K, int Nc) {
    long idx = (long)blockIdx.x * 256 + threadIdx.x;
    if (idx >= (long)K * Nc) return;
    int  kk = (int)(idx % K);
    long n  = idx / K;
    Wt[n * K + kk] = (_Float16)W[(size_t)kk * Nc + n];
}

// ---------------- LN1: x[B,C,300] -> h16[B,400,C] (pad rows zero) ----------------
__global__ __launch_bounds__(256) void ln1_kernel(const float* __restrict__ x,
                                                  const float* __restrict__ sc,
                                                  const float* __restrict__ bi,
                                                  _Float16* __restrict__ h16) {
    int wid = threadIdx.x >> 5, lane = threadIdx.x & 31;
    long row = (long)blockIdx.x * 8 + wid;            // 0..51199
    int b = (int)(row / NTOK), tok = (int)(row % NTOK);
    _Float16* out = h16 + (size_t)row * CC;
    if (tok >= LTOK) {
        for (int c = lane; c < CC; c += 32) out[c] = (_Float16)0.f;
        return;
    }
    const float* xb = x + (size_t)b * CC * LTOK + tok;
    float v[12], sum = 0.f;
    #pragma unroll
    for (int i = 0; i < 12; ++i) { v[i] = xb[(size_t)(lane + i * 32) * LTOK]; sum += v[i]; }
    float mean = wave_sum(sum) * (1.0f / CC);
    float var = 0.f;
    #pragma unroll
    for (int i = 0; i < 12; ++i) { float d = v[i] - mean; var += d * d; }
    var = wave_sum(var) * (1.0f / CC);
    float rstd = rsqrtf(var + 1e-5f);
    #pragma unroll
    for (int i = 0; i < 12; ++i) {
        int c = lane + i * 32;
        out[c] = (_Float16)((v[i] - mean) * rstd * sc[c] + bi[c]);
    }
}

// ---------------- zero v^T padding tokens 400..415 ----------------
__global__ void vpad_kernel(_Float16* __restrict__ vT) {
    long idx = (long)blockIdx.x * 256 + threadIdx.x;   // BB*NHH*HD*16
    if (idx >= (long)BB * NHH * HD * 16) return;
    int p = (int)(idx & 15);
    long t = idx >> 4;                                  // bh*64+hd
    vT[t * NTOKP + NTOK + p] = (_Float16)0.f;
}

// ---------------- QKV GEMM: h16[51200,384] x qkvT[1152,384], NB=4 ----------------
__global__ __launch_bounds__(256) void gemm_qkv_kernel(const _Float16* __restrict__ A,
                                                       const _Float16* __restrict__ Bt,
                                                       const float* __restrict__ qb,
                                                       _Float16* __restrict__ q,
                                                       _Float16* __restrict__ k,
                                                       _Float16* __restrict__ v) {
    const int K = CC, TG = 18;                          // 1152 / 64
    int wave = threadIdx.x >> 5, lane = threadIdx.x & 31;
    long tile = (long)blockIdx.x * 8 + wave;
    if (tile >= (long)3200 * TG) return;
    int tm = (int)(tile / TG), tg = (int)(tile % TG);
    int rsel = lane & 15, koff = (lane >> 4) << 3;
    v8f acc[4] = {};
    gemm_k_loop_nb<4>(A + (size_t)(tm * 16 + rsel) * K + koff,
                      Bt + (size_t)(tg * 64 + rsel) * K + koff,
                      (long)16 * K, K, acc);
    int mloc = (lane >> 4) << 3, nloc = lane & 15;
    #pragma unroll
    for (int j = 0; j < 4; ++j) {
        int n = tg * 64 + j * 16 + nloc;
        int which = n / CC, c = n % CC, hh = c >> 6, hd = c & 63;
        float bias = qb[n];
        #pragma unroll
        for (int r = 0; r < 8; ++r) {
            long m = (long)tm * 16 + mloc + r;
            long b = m / NTOK, tok = m % NTOK;
            float val = acc[j][r] + bias;
            size_t bh = (size_t)b * NHH + hh;
            if (which == 0)      q[(bh * NTOK + tok) * HD + hd] = (_Float16)(val * 0.125f);
            else if (which == 1) k[(bh * NTOK + tok) * HD + hd] = (_Float16)val;
            else                 v[(bh * HD + hd) * NTOKP + tok] = (_Float16)val;
        }
    }
}

// ---------------- Attention: one wave per (b, head, 16-query tile) ----------------
__global__ __launch_bounds__(32) void attn_kernel(const _Float16* __restrict__ qf,
                                                  const _Float16* __restrict__ kf,
                                                  const _Float16* __restrict__ vT,
                                                  const float* __restrict__ bias_table,
                                                  _Float16* __restrict__ o) {
    __shared__ __align__(16) float     sS[16 * NTOK];
    __shared__ __align__(16) _Float16  sP[16 * NTOKP];
    int lane = threadIdx.x;
    int bid = blockIdx.x;
    int qt = bid % 25;
    int bh = bid / 25;                  // b*NHH + h
    int h  = bh % NHH;
    size_t bimg = (size_t)(bh / NHH);

    const _Float16* qb = qf + (size_t)bh * NTOK * HD;
    const _Float16* kb = kf + (size_t)bh * NTOK * HD;
    const _Float16* vb = vT + (size_t)bh * HD * NTOKP;

    int rsel = lane & 15;
    int koff = (lane >> 4) << 3;
    int mloc = (lane >> 4) << 3;

    const _Float16* pq = qb + (size_t)(qt * 16 + rsel) * HD + koff;
    v16h a0 = load_frag16(pq);
    v16h a1 = load_frag16(pq + 32);

    // scores + bias into LDS
    for (int j = 0; j < 25; ++j) {
        const _Float16* pk = kb + (size_t)(j * 16 + rsel) * HD + koff;
        v16h b0 = load_frag16(pk);
        v16h b1 = load_frag16(pk + 32);
        v8f acc = {};
        acc = __builtin_amdgcn_wmma_f32_16x16x32_f16(false, a0, false, b0, (short)0, acc, false, false);
        acc = __builtin_amdgcn_wmma_f32_16x16x32_f16(false, a1, false, b1, (short)0, acc, false, false);
        int n = lane & 15;
        int ktok = j * 16 + n;
        int ky = ktok / WS, kx = ktok % WS;
        #pragma unroll
        for (int r = 0; r < 8; ++r) {
            int m = mloc + r;
            int qtok = qt * 16 + m;
            int qy = qtok / WS, qx = qtok % WS;
            int ridx = (qy - ky + WS - 1) * (2 * WS - 1) + (qx - kx + WS - 1);
            sS[m * NTOK + ktok] = acc[r] + bias_table[ridx * NHH + h];
        }
    }
    __syncthreads();

    // softmax over 400 keys, 2 lanes per row
    {
        int row = lane & 15;
        int halfsel = lane >> 4;
        int start = halfsel * 200;
        float mx = -3.4e38f;
        for (int t = 0; t < 200; ++t) mx = fmaxf(mx, sS[row * NTOK + start + t]);
        mx = fmaxf(mx, __shfl_xor(mx, 16, 32));
        float sum = 0.f;
        for (int t = 0; t < 200; ++t) sum += __expf(sS[row * NTOK + start + t] - mx);
        sum += __shfl_xor(sum, 16, 32);
        float inv = 1.0f / sum;
        for (int t = 0; t < 200; ++t)
            sP[row * NTOKP + start + t] = (_Float16)(__expf(sS[row * NTOK + start + t] - mx) * inv);
        if (halfsel == 1)
            for (int t = NTOK; t < NTOKP; ++t) sP[row * NTOKP + t] = (_Float16)0.f;
    }
    __syncthreads();

    // O = P @ V  (K = 416, 4 tiles of HD)
    const _Float16* pp = &sP[(size_t)rsel * NTOKP + koff];
    for (int nt = 0; nt < 4; ++nt) {
        const _Float16* pv = vb + (size_t)(nt * 16 + rsel) * NTOKP + koff;
        v8f acc = {};
        acc = gemm_k_loop(pp, pv, NTOKP, acc);
        int n = lane & 15;
        #pragma unroll
        for (int r = 0; r < 8; ++r) {
            int m = mloc + r;
            size_t tok = (size_t)qt * 16 + m;
            o[(bimg * NTOK + tok) * CC + h * HD + nt * 16 + n] = (_Float16)acc[r];
        }
    }
}

// ---------------- proj GEMM + residual -> t32[B,300,C], NB=4 ----------------
__global__ __launch_bounds__(256) void gemm_proj_kernel(const _Float16* __restrict__ A,
                                                        const _Float16* __restrict__ Bt,
                                                        const float* __restrict__ pbias,
                                                        const float* __restrict__ x,
                                                        float* __restrict__ t32) {
    const int K = CC, TG = 6;                           // 384 / 64
    int wave = threadIdx.x >> 5, lane = threadIdx.x & 31;
    long tile = (long)blockIdx.x * 8 + wave;
    if (tile >= (long)3200 * TG) return;
    int tm = (int)(tile / TG), tg = (int)(tile % TG);
    int rsel = lane & 15, koff = (lane >> 4) << 3;
    v8f acc[4] = {};
    gemm_k_loop_nb<4>(A + (size_t)(tm * 16 + rsel) * K + koff,
                      Bt + (size_t)(tg * 64 + rsel) * K + koff,
                      (long)16 * K, K, acc);
    int mloc = (lane >> 4) << 3, nloc = lane & 15;
    #pragma unroll
    for (int j = 0; j < 4; ++j) {
        int n = tg * 64 + j * 16 + nloc;
        float bias = pbias[n];
        #pragma unroll
        for (int r = 0; r < 8; ++r) {
            long m = (long)tm * 16 + mloc + r;
            long b = m / NTOK, tok = m % NTOK;
            if (tok < LTOK) {
                float val = acc[j][r] + bias + x[((size_t)b * CC + n) * LTOK + tok];
                t32[((size_t)b * LTOK + tok) * CC + n] = val;
            }
        }
    }
}

// ---------------- LN2: t32 rows -> m16 ----------------
__global__ __launch_bounds__(256) void ln2_kernel(const float* __restrict__ t32,
                                                  const float* __restrict__ sc,
                                                  const float* __restrict__ bi,
                                                  _Float16* __restrict__ m16) {
    int wid = threadIdx.x >> 5, lane = threadIdx.x & 31;
    long row = (long)blockIdx.x * 8 + wid;           // 0..38399
    const float* tr = t32 + (size_t)row * CC;
    _Float16* out = m16 + (size_t)row * CC;
    float v[12], sum = 0.f;
    #pragma unroll
    for (int i = 0; i < 12; ++i) { v[i] = tr[lane + i * 32]; sum += v[i]; }
    float mean = wave_sum(sum) * (1.0f / CC);
    float var = 0.f;
    #pragma unroll
    for (int i = 0; i < 12; ++i) { float d = v[i] - mean; var += d * d; }
    var = wave_sum(var) * (1.0f / CC);
    float rstd = rsqrtf(var + 1e-5f);
    #pragma unroll
    for (int i = 0; i < 12; ++i) {
        int c = lane + i * 32;
        out[c] = (_Float16)((v[i] - mean) * rstd * sc[c] + bi[c]);
    }
}

// ---------------- fc1 GEMM + GELU, NB=4 ----------------
__global__ __launch_bounds__(256) void gemm_fc1_kernel(const _Float16* __restrict__ A,
                                                       const _Float16* __restrict__ Bt,
                                                       const float* __restrict__ fb,
                                                       _Float16* __restrict__ g16) {
    const int K = CC, TG = 24;                         // 1536 / 64
    int wave = threadIdx.x >> 5, lane = threadIdx.x & 31;
    long tile = (long)blockIdx.x * 8 + wave;
    if (tile >= (long)2400 * TG) return;
    int tm = (int)(tile / TG), tg = (int)(tile % TG);
    int rsel = lane & 15, koff = (lane >> 4) << 3;
    v8f acc[4] = {};
    gemm_k_loop_nb<4>(A + (size_t)(tm * 16 + rsel) * K + koff,
                      Bt + (size_t)(tg * 64 + rsel) * K + koff,
                      (long)16 * K, K, acc);
    int mloc = (lane >> 4) << 3, nloc = lane & 15;
    #pragma unroll
    for (int j = 0; j < 4; ++j) {
        int n = tg * 64 + j * 16 + nloc;
        float bias = fb[n];
        #pragma unroll
        for (int r = 0; r < 8; ++r) {
            long m = (long)tm * 16 + mloc + r;
            float val = acc[j][r] + bias;
            val = 0.5f * val * (1.0f + erff(val * 0.70710678f));
            g16[(size_t)m * 1536 + n] = (_Float16)val;
        }
    }
}

// ---------------- fc2 GEMM + residual -> t2_16, NB=4 ----------------
__global__ __launch_bounds__(256) void gemm_fc2_kernel(const _Float16* __restrict__ A,
                                                       const _Float16* __restrict__ Bt,
                                                       const float* __restrict__ fb,
                                                       const float* __restrict__ t32,
                                                       _Float16* __restrict__ t2) {
    const int K = 1536, TG = 6;                        // 384 / 64
    int wave = threadIdx.x >> 5, lane = threadIdx.x & 31;
    long tile = (long)blockIdx.x * 8 + wave;
    if (tile >= (long)2400 * TG) return;
    int tm = (int)(tile / TG), tg = (int)(tile % TG);
    int rsel = lane & 15, koff = (lane >> 4) << 3;
    v8f acc[4] = {};
    gemm_k_loop_nb<4>(A + (size_t)(tm * 16 + rsel) * K + koff,
                      Bt + (size_t)(tg * 64 + rsel) * K + koff,
                      (long)16 * K, K, acc);
    int mloc = (lane >> 4) << 3, nloc = lane & 15;
    #pragma unroll
    for (int j = 0; j < 4; ++j) {
        int n = tg * 64 + j * 16 + nloc;
        float bias = fb[n];
        #pragma unroll
        for (int r = 0; r < 8; ++r) {
            long m = (long)tm * 16 + mloc + r;
            float val = acc[j][r] + bias + t32[(size_t)m * CC + n];
            t2[(size_t)m * CC + n] = (_Float16)val;
        }
    }
}

// ---------------- out GEMM + ReLU -> y32[B,192,300], NB=4 ----------------
__global__ __launch_bounds__(256) void gemm_out_kernel(const _Float16* __restrict__ A,
                                                       const _Float16* __restrict__ Bt,
                                                       const float* __restrict__ ob,
                                                       float* __restrict__ y32) {
    const int K = CC, TG = 3;                           // 192 / 64
    int wave = threadIdx.x >> 5, lane = threadIdx.x & 31;
    long tile = (long)blockIdx.x * 8 + wave;
    if (tile >= (long)2400 * TG) return;
    int tm = (int)(tile / TG), tg = (int)(tile % TG);
    int rsel = lane & 15, koff = (lane >> 4) << 3;
    v8f acc[4] = {};
    gemm_k_loop_nb<4>(A + (size_t)(tm * 16 + rsel) * K + koff,
                      Bt + (size_t)(tg * 64 + rsel) * K + koff,
                      (long)16 * K, K, acc);
    int mloc = (lane >> 4) << 3, nloc = lane & 15;
    #pragma unroll
    for (int j = 0; j < 4; ++j) {
        int n = tg * 64 + j * 16 + nloc;
        float bias = ob[n];
        #pragma unroll
        for (int r = 0; r < 8; ++r) {
            long m = (long)tm * 16 + mloc + r;
            long b = m / LTOK, tok = m % LTOK;
            float val = fmaxf(acc[j][r] + bias, 0.0f);
            y32[((size_t)b * OUTD + n) * LTOK + tok] = val;
        }
    }
}

// ---------------- bilinear 2x resize [B,192,15,20] -> [B,192,30,40] ----------------
__global__ void resize_kernel(const float* __restrict__ y, float* __restrict__ out) {
    long idx = (long)blockIdx.x * 256 + threadIdx.x;    // BB*OUTD*OH*OW
    if (idx >= (long)BB * OUTD * OH * OW) return;
    int ox = (int)(idx % OW); long t = idx / OW;
    int oy = (int)(t % OH);   t /= OH;
    int c  = (int)(t % OUTD);
    int b  = (int)(t / OUTD);
    float sy = fmaxf((oy + 0.5f) * 0.5f - 0.5f, 0.0f);
    float sx = fmaxf((ox + 0.5f) * 0.5f - 0.5f, 0.0f);
    int y0 = (int)floorf(sy), x0 = (int)floorf(sx);
    float wy = sy - y0, wx = sx - x0;
    int y1 = min(y0 + 1, HH - 1), x1 = min(x0 + 1, WW - 1);
    y0 = min(y0, HH - 1); x0 = min(x0, WW - 1);
    const float* p = y + ((size_t)b * OUTD + c) * LTOK;
    float v00 = p[y0 * WW + x0], v01 = p[y0 * WW + x1];
    float v10 = p[y1 * WW + x0], v11 = p[y1 * WW + x1];
    out[idx] = v00 * (1 - wy) * (1 - wx) + v01 * (1 - wy) * wx
             + v10 * wy * (1 - wx)       + v11 * wy * wx;
}

extern "C" void kernel_launch(void* const* d_in, const int* in_sizes, int n_in,
                              void* d_out, int out_size, void* d_ws, size_t ws_size,
                              hipStream_t stream) {
    const float* x       = (const float*)d_in[0];
    const float* ln1_s   = (const float*)d_in[1];
    const float* ln1_b   = (const float*)d_in[2];
    const float* qkv_w   = (const float*)d_in[3];
    const float* qkv_b   = (const float*)d_in[4];
    const float* btab    = (const float*)d_in[5];
    const float* proj_w  = (const float*)d_in[6];
    const float* proj_b  = (const float*)d_in[7];
    const float* ln2_s   = (const float*)d_in[8];
    const float* ln2_b   = (const float*)d_in[9];
    const float* fc1_w   = (const float*)d_in[10];
    const float* fc1_b   = (const float*)d_in[11];
    const float* fc2_w   = (const float*)d_in[12];
    const float* fc2_b   = (const float*)d_in[13];
    const float* out_w   = (const float*)d_in[14];
    const float* out_b   = (const float*)d_in[15];

    char* base = (char*)d_ws;
    size_t off = 0;
    auto take = [&](size_t bytes) -> void* {
        void* r = base + off;
        off += (bytes + 255) & ~(size_t)255;
        return r;
    };
    _Float16* qkvT = (_Float16*)take((size_t)1152 * 384 * 2);
    _Float16* projT= (_Float16*)take((size_t)384 * 384 * 2);
    _Float16* fc1T = (_Float16*)take((size_t)1536 * 384 * 2);
    _Float16* fc2T = (_Float16*)take((size_t)384 * 1536 * 2);
    _Float16* outT = (_Float16*)take((size_t)192 * 384 * 2);
    _Float16* h16  = (_Float16*)take((size_t)BB * NTOK * CC * 2);           // reused: t2_16
    _Float16* q16  = (_Float16*)take((size_t)BB * NHH * NTOK * HD * 2);     // reuse start: g16
    _Float16* k16  = (_Float16*)take((size_t)BB * NHH * NTOK * HD * 2);
    _Float16* vT16 = (_Float16*)take((size_t)BB * NHH * HD * NTOKP * 2);
    _Float16* o16  = (_Float16*)take((size_t)BB * NTOK * CC * 2);           // reused: m16, y32
    float*    t32  = (float*)   take((size_t)BB * LTOK * CC * 4);
    _Float16* g16   = q16;     // 38400*1536*2 fits inside q+k+v region
    _Float16* t2_16 = h16;
    _Float16* m16   = o16;
    float*    y32   = (float*)o16;

    // weight conversion / transpose
    wconv_kernel<<<(384 * 1152 + 255) / 256, 256, 0, stream>>>(qkv_w, qkvT, 384, 1152);
    wconv_kernel<<<(384 * 384  + 255) / 256, 256, 0, stream>>>(proj_w, projT, 384, 384);
    wconv_kernel<<<(384 * 1536 + 255) / 256, 256, 0, stream>>>(fc1_w, fc1T, 384, 1536);
    wconv_kernel<<<(1536 * 384 + 255) / 256, 256, 0, stream>>>(fc2_w, fc2T, 1536, 384);
    wconv_kernel<<<(384 * 192  + 255) / 256, 256, 0, stream>>>(out_w, outT, 384, 192);

    ln1_kernel<<<BB * NTOK / 8, 256, 0, stream>>>(x, ln1_s, ln1_b, h16);
    vpad_kernel<<<(BB * NHH * HD * 16 + 255) / 256, 256, 0, stream>>>(vT16);

    gemm_qkv_kernel<<<3200 * 18 / 8, 256, 0, stream>>>(h16, qkvT, qkv_b, q16, k16, vT16);
    attn_kernel<<<BB * NHH * 25, 32, 0, stream>>>(q16, k16, vT16, btab, o16);
    gemm_proj_kernel<<<3200 * 6 / 8, 256, 0, stream>>>(o16, projT, proj_b, x, t32);
    ln2_kernel<<<BB * LTOK / 8, 256, 0, stream>>>(t32, ln2_s, ln2_b, m16);
    gemm_fc1_kernel<<<2400 * 24 / 8, 256, 0, stream>>>(m16, fc1T, fc1_b, g16);
    gemm_fc2_kernel<<<2400 * 6 / 8, 256, 0, stream>>>(g16, fc2T, fc2_b, t32, t2_16);
    gemm_out_kernel<<<2400 * 3 / 8, 256, 0, stream>>>(t2_16, outT, out_b, y32);
    resize_kernel<<<(BB * OUTD * OH * OW + 255) / 256, 256, 0, stream>>>(y32, (float*)d_out);
}
